// UNetSphericalHealpixResidualShort2Levels_67869073211453
// MI455X (gfx1250) — matrix operand order repaired and checked by reading, not compiled, not run on hardware
//
#include <hip/hip_runtime.h>
#include <hip/hip_bf16.h>
#include <stdint.h>

#define BATCH 32
#define NN    3072
#define NN2   768

typedef __attribute__((ext_vector_type(16))) __bf16 v16bf;
typedef __attribute__((ext_vector_type(8)))  float  v8f;

__device__ __forceinline__ unsigned short f2bf(float f) {
  union { float f; unsigned u; } x; x.f = f;
  unsigned r = 0x7FFFu + ((x.u >> 16) & 1u);     // round-to-nearest-even
  return (unsigned short)((x.u + r) >> 16);
}
__device__ __forceinline__ float bf2f(unsigned short h) {
  union { unsigned u; float f; } x; x.u = ((unsigned)h) << 16;
  return x.f;
}

union ABf { v16bf v; uint4 q[2]; };

// A-tile load per CDNA5 16-bit A layout: lane half g gets K chunks [8g,8g+8) and
// [16+8g,24+8g) as two contiguous 16B loads.
__device__ __forceinline__ void loadA(ABf& a, const unsigned short* __restrict__ arow,
                                      int k, int g) {
  a.q[0] = *reinterpret_cast<const uint4*>(arow + k + 8 * g);
  a.q[1] = *reinterpret_cast<const uint4*>(arow + k + 16 + 8 * g);
}
// B-tile load: lane&15 = column, K chunks [8g,8g+8) and [16+8g,24+8g) (same as A).
__device__ __forceinline__ void loadB(ABf& b, const unsigned short* __restrict__ brow,
                                      int k, int g) {
  const unsigned short* p = brow + k + 8 * g;
  b.q[0] = *reinterpret_cast<const uint4*>(p);
  b.q[1] = *reinterpret_cast<const uint4*>(p + 16);
}

// ---------------------------------------------------------------------------
// Generic bf16 WMMA GEMM:
//   D[m][n] = alpha * sum_k A[m][k]*Bt[n][k] + beta*Z[m][n] + bias[n]
// A row-major [M][K] bf16; Bt row-major [Ncols_pad][K] bf16 (i.e. B^T).
// Z read from f32 (Zf) or bf16-natural (Zh). Outputs: f32 D, bf16 natural
// Dnat (same indexing as D), bf16 transposed Dt[n][m] — any may be null.
// One wave per block computing a (16*MT) x (16*NT) tile; K loop is a two-stage
// software pipeline (ping-pong registers) so VMEM latency overlaps the wmmas.
// ---------------------------------------------------------------------------
template <int MT, int NT>
__global__ __launch_bounds__(32) void k_gemm(
    const unsigned short* __restrict__ A,  long long sA, int lda,
    const unsigned short* __restrict__ Bt, long long sB, int ldb,
    const float* __restrict__ Zf, const unsigned short* __restrict__ Zh,
    long long sZ, int ldz,
    float alpha, float beta,
    const float* __restrict__ bias,
    float* __restrict__ D, long long sD, int ldd,
    unsigned short* __restrict__ Dnat,
    unsigned short* __restrict__ Dt, long long sDt, int lddt,
    int Ncols, int Kdim)
{
  const int lane = threadIdx.x;
  const int g = lane >> 4, h = lane & 15;
  const int m0 = blockIdx.x * (16 * MT);
  const int n0 = blockIdx.y * (16 * NT);
  const int b  = blockIdx.z;

  const unsigned short* arow[MT];
#pragma unroll
  for (int mt = 0; mt < MT; ++mt)
    arow[mt] = A + (long long)b * sA + (long long)(m0 + mt * 16 + h) * lda;
  const unsigned short* brow[NT];
#pragma unroll
  for (int t = 0; t < NT; ++t)
    brow[t] = Bt + (long long)b * sB + (long long)(n0 + t * 16 + h) * ldb;

  const v8f vzero = {0.f,0.f,0.f,0.f,0.f,0.f,0.f,0.f};
  v8f acc[MT][NT];
#pragma unroll
  for (int mt = 0; mt < MT; ++mt)
#pragma unroll
    for (int t = 0; t < NT; ++t) acc[mt][t] = vzero;

  ABf a0[MT], a1[MT], b0[NT], b1[NT];
#pragma unroll
  for (int mt = 0; mt < MT; ++mt) loadA(a0[mt], arow[mt], 0, g);
#pragma unroll
  for (int t = 0; t < NT; ++t)    loadB(b0[t], brow[t], 0, g);

  int k0 = 0;
  for (; k0 + 64 <= Kdim; k0 += 64) {
    // stage 1: issue loads for k0+32, consume k0
#pragma unroll
    for (int mt = 0; mt < MT; ++mt) loadA(a1[mt], arow[mt], k0 + 32, g);
#pragma unroll
    for (int t = 0; t < NT; ++t)    loadB(b1[t], brow[t], k0 + 32, g);
#pragma unroll
    for (int mt = 0; mt < MT; ++mt)
#pragma unroll
      for (int t = 0; t < NT; ++t)
        acc[mt][t] = __builtin_amdgcn_wmma_f32_16x16x32_bf16(
            false, a0[mt].v, false, b0[t].v, (short)0, acc[mt][t], false, false);
    // stage 2: issue loads for k0+64 (if any), consume k0+32
    if (k0 + 64 < Kdim) {
#pragma unroll
      for (int mt = 0; mt < MT; ++mt) loadA(a0[mt], arow[mt], k0 + 64, g);
#pragma unroll
      for (int t = 0; t < NT; ++t)    loadB(b0[t], brow[t], k0 + 64, g);
    }
#pragma unroll
    for (int mt = 0; mt < MT; ++mt)
#pragma unroll
      for (int t = 0; t < NT; ++t)
        acc[mt][t] = __builtin_amdgcn_wmma_f32_16x16x32_bf16(
            false, a1[mt].v, false, b1[t].v, (short)0, acc[mt][t], false, false);
  }
  if (k0 < Kdim) {               // tail 32 (Kdim % 64 == 32), tiles already in a0/b0
#pragma unroll
    for (int mt = 0; mt < MT; ++mt)
#pragma unroll
      for (int t = 0; t < NT; ++t)
        acc[mt][t] = __builtin_amdgcn_wmma_f32_16x16x32_bf16(
            false, a0[mt].v, false, b0[t].v, (short)0, acc[mt][t], false, false);
  }

#pragma unroll
  for (int mt = 0; mt < MT; ++mt) {
#pragma unroll
    for (int t = 0; t < NT; ++t) {
      int n = n0 + t * 16 + h;
      if (n >= Ncols) continue;
      float bv = bias ? bias[n] : 0.0f;
#pragma unroll
      for (int r = 0; r < 8; ++r) {
        int m = m0 + mt * 16 + r + 8 * g;     // C/D layout: m = VGPR + 8*(lane/16)
        float v = alpha * acc[mt][t][r] + bv;
        long long zi = (long long)b * sZ + (long long)m * ldz + n;
        if (Zf)      v += beta * Zf[zi];
        else if (Zh) v += beta * bf2f(Zh[zi]);
        long long di = (long long)b * sD + (long long)m * ldd + n;
        if (D)    D[di]    = v;
        unsigned short hv = f2bf(v);
        if (Dnat) Dnat[di] = hv;
        if (Dt)   Dt[(long long)b * sDt + (long long)n * lddt + m] = hv;
      }
    }
  }
}

// --------------------------- elementwise kernels ---------------------------

__global__ void k_f32_to_bf16(const float* __restrict__ X,
                              unsigned short* __restrict__ Y, long long n) {
  long long i = (long long)blockIdx.x * 256 + threadIdx.x;
  if (i < n) Y[i] = f2bf(X[i]);
}

// W[3][Cin][Cout] f32 -> Wt[3][Pad][Cin] bf16 (transposed per k, zero padded)
__global__ void k_cheb_w(const float* __restrict__ W, unsigned short* __restrict__ Wt,
                         int Cin, int Cout, int Pad) {
  long long i = (long long)blockIdx.x * 256 + threadIdx.x;
  long long tot = 3LL * Pad * Cin;
  if (i >= tot) return;
  int c = (int)(i % Cin);
  int o = (int)((i / Cin) % Pad);
  int k = (int)(i / ((long long)Cin * Pad));
  float v = (o < Cout) ? W[((long long)k * Cin + c) * Cout + o] : 0.0f;
  Wt[i] = f2bf(v);
}

// f32 [B][nodes][C] -> bf16 natural + bf16 transposed [B][C][nodes]
__global__ void k_f32_to_natT(const float* __restrict__ X, long long total,
                              int C, int nodes,
                              unsigned short* __restrict__ nat,
                              unsigned short* __restrict__ T) {
  long long i = (long long)blockIdx.x * 256 + threadIdx.x;
  if (i >= total) return;
  int c = (int)(i % C);
  long long r = i / C;
  int node = (int)(r % nodes);
  int b = (int)(r / nodes);
  unsigned short hv = f2bf(X[i]);
  nat[i] = hv;
  T[((long long)b * C + c) * nodes + node] = hv;
}

// per-channel sum + sumsq over rows (one block per channel)
__global__ void k_bn_reduce(const float* __restrict__ Y, int rows, int C,
                            float* __restrict__ sums) {
  __shared__ float s1[256];
  __shared__ float s2[256];
  int c = blockIdx.x, t = threadIdx.x;
  float a = 0.f, b = 0.f;
  for (int r = t; r < rows; r += 256) {
    float v = Y[(long long)r * C + c];
    a += v; b += v * v;
  }
  s1[t] = a; s2[t] = b; __syncthreads();
  for (int off = 128; off > 0; off >>= 1) {
    if (t < off) { s1[t] += s1[t + off]; s2[t] += s2[t + off]; }
    __syncthreads();
  }
  if (t == 0) { sums[c] = s1[0]; sums[C + c] = s2[0]; }
}

// BN + ReLU, emit bf16 natural + transposed
__global__ void k_bn_apply(const float* __restrict__ Y, const float* __restrict__ sums,
                           const float* __restrict__ gamma, const float* __restrict__ beta,
                           int rows, int C, int nodes,
                           unsigned short* __restrict__ Onat,
                           unsigned short* __restrict__ Ot) {
  long long i = (long long)blockIdx.x * 256 + threadIdx.x;
  long long total = (long long)rows * C;
  if (i >= total) return;
  int c = (int)(i % C);
  long long r = i / C;
  float cnt = (float)rows;
  float mu = sums[c] / cnt;
  float var = sums[C + c] / cnt - mu * mu;
  float rs = rsqrtf(var + 1e-5f);
  float v = (Y[i] - mu) * rs * gamma[c] + beta[c];
  v = fmaxf(v, 0.0f);
  unsigned short hv = f2bf(v);
  Onat[i] = hv;
  int node = (int)(r % nodes);
  int b = (int)(r / nodes);
  Ot[((long long)b * C + c) * nodes + node] = hv;
}

// max-pool /4 with argmax (first-max tie-break like jnp.argmax)
__global__ void k_pool(const float* __restrict__ X, int nodes, int C,
                       unsigned short* __restrict__ pnat,
                       unsigned short* __restrict__ pT,
                       int* __restrict__ idx, long long total) {
  long long i = (long long)blockIdx.x * 256 + threadIdx.x;
  if (i >= total) return;
  int c = (int)(i % C);
  long long t = i / C;
  int nout = nodes / 4;
  int j = (int)(t % nout);
  int b = (int)(t / nout);
  const float* base = X + ((long long)b * nodes + (long long)j * 4) * C + c;
  float best = base[0]; int bi = 0;
  for (int q = 1; q < 4; ++q) {
    float v = base[(long long)q * C];
    if (v > best) { best = v; bi = q; }
  }
  idx[i] = j * 4 + bi;
  unsigned short hv = f2bf(best);
  pnat[i] = hv;
  pT[((long long)b * C + c) * nout + j] = hv;
}

// cat bf16: channels [0,128) = 0 (unpool target), [128,256) = e1
__global__ void k_cat_init(const float* __restrict__ e1, int nodes,
                           unsigned short* __restrict__ cnat,
                           unsigned short* __restrict__ cT, long long total) {
  long long i = (long long)blockIdx.x * 256 + threadIdx.x;
  if (i >= total) return;
  int c = (int)(i % 256);
  long long t = i / 256;
  int node = (int)(t % nodes);
  int b = (int)(t / nodes);
  float v = (c >= 128) ? e1[((long long)b * nodes + node) * 128 + (c - 128)] : 0.0f;
  unsigned short hv = f2bf(v);
  cnat[i] = hv;
  cT[((long long)b * 256 + c) * nodes + node] = hv;
}

// scatter e2 (half-res) into cat channels [0,128) at argmax positions
__global__ void k_scatter(const float* __restrict__ e2, const int* __restrict__ idx,
                          int nodes, unsigned short* __restrict__ cnat,
                          unsigned short* __restrict__ cT, long long total) {
  long long i = (long long)blockIdx.x * 256 + threadIdx.x;
  if (i >= total) return;
  int c = (int)(i % 128);
  long long t = i / 128;
  int b = (int)(t / (nodes / 4));
  int node = idx[i];
  unsigned short hv = f2bf(e2[i]);
  cnat[((long long)b * nodes + node) * 256 + c] = hv;
  cT[((long long)b * 256 + c) * nodes + node] = hv;
}

// ---------------------------------------------------------------------------

extern "C" void kernel_launch(void* const* d_in, const int* in_sizes, int n_in,
                              void* d_out, int out_size, void* d_ws, size_t ws_size,
                              hipStream_t stream) {
  (void)in_sizes; (void)n_in; (void)out_size;

  const float* x    = (const float*)d_in[0];
  const float* L0   = (const float*)d_in[1];
  const float* L1   = (const float*)d_in[2];
  const float* w11  = (const float*)d_in[3];  const float* b11  = (const float*)d_in[4];
  const float* g11  = (const float*)d_in[5];  const float* be11 = (const float*)d_in[6];
  const float* w13  = (const float*)d_in[7];  const float* b13  = (const float*)d_in[8];
  const float* g13  = (const float*)d_in[9];  const float* be13 = (const float*)d_in[10];
  const float* wr1  = (const float*)d_in[11]; const float* br1  = (const float*)d_in[12];
  const float* w21  = (const float*)d_in[13]; const float* b21  = (const float*)d_in[14];
  const float* g21  = (const float*)d_in[15]; const float* be21 = (const float*)d_in[16];
  const float* w23  = (const float*)d_in[17]; const float* b23  = (const float*)d_in[18];
  const float* g23  = (const float*)d_in[19]; const float* be23 = (const float*)d_in[20];
  const float* wr2  = (const float*)d_in[21]; const float* br2  = (const float*)d_in[22];
  const float* uw11 = (const float*)d_in[23]; const float* ub11 = (const float*)d_in[24];
  const float* ug11 = (const float*)d_in[25]; const float* ube11= (const float*)d_in[26];
  const float* uw12 = (const float*)d_in[27]; const float* ub12 = (const float*)d_in[28];
  const float* ug12 = (const float*)d_in[29]; const float* ube12= (const float*)d_in[30];
  const float* uwr1 = (const float*)d_in[31]; const float* ubr1 = (const float*)d_in[32];
  const float* uw13 = (const float*)d_in[33]; const float* ub13 = (const float*)d_in[34];

  // ----- workspace bump allocator (256B aligned) -----
  size_t off = 0;
  auto alloc = [&](size_t bytes) -> char* {
    off = (off + 255) & ~(size_t)255;
    char* p = (char*)d_ws + off;
    off += bytes;
    return p;
  };
  unsigned short* L0h   = (unsigned short*)alloc((size_t)NN * NN * 2);
  unsigned short* L1h   = (unsigned short*)alloc((size_t)NN2 * NN2 * 2);
  unsigned short* w11t  = (unsigned short*)alloc((size_t)3 * 64  * 32  * 2);
  unsigned short* w13t  = (unsigned short*)alloc((size_t)3 * 128 * 64  * 2);
  unsigned short* w21t  = (unsigned short*)alloc((size_t)3 * 192 * 128 * 2);
  unsigned short* w23t  = (unsigned short*)alloc((size_t)3 * 128 * 192 * 2);
  unsigned short* uw11t = (unsigned short*)alloc((size_t)3 * 128 * 256 * 2);
  unsigned short* uw12t = (unsigned short*)alloc((size_t)3 * 64  * 128 * 2);
  unsigned short* uw13t = (unsigned short*)alloc((size_t)3 * 16  * 64  * 2);
  unsigned short* wr1h  = (unsigned short*)alloc((size_t)128 * 32  * 2);
  unsigned short* wr2h  = (unsigned short*)alloc((size_t)128 * 128 * 2);
  unsigned short* uwr1h = (unsigned short*)alloc((size_t)64  * 256 * 2);
  unsigned short* xnat  = (unsigned short*)alloc((size_t)BATCH * NN * 32 * 2);
  unsigned short* xT    = (unsigned short*)alloc((size_t)BATCH * NN * 32 * 2);
  unsigned short* b1nat = (unsigned short*)alloc((size_t)BATCH * NN * 256 * 2);
  unsigned short* b1T   = (unsigned short*)alloc((size_t)BATCH * NN * 256 * 2);
  unsigned short* b2nat = (unsigned short*)alloc((size_t)BATCH * NN * 256 * 2);
  float*          pre   = (float*)alloc((size_t)BATCH * NN * 128 * 4);
  unsigned short* P0nat = (unsigned short*)alloc((size_t)BATCH * NN * 128 * 2);
  unsigned short* P0T   = (unsigned short*)alloc((size_t)BATCH * NN * 128 * 2);
  unsigned short* P1nat = (unsigned short*)alloc((size_t)BATCH * NN * 128 * 2);
  unsigned short* P1T   = (unsigned short*)alloc((size_t)BATCH * NN * 128 * 2);
  float*          e1f   = (float*)alloc((size_t)BATCH * NN * 128 * 4);
  int*            idx   = (int*)alloc((size_t)BATCH * NN2 * 128 * 4);
  float*          e2f   = (float*)alloc((size_t)BATCH * NN2 * 128 * 4);
  unsigned short* catnat= (unsigned short*)alloc((size_t)BATCH * NN * 256 * 2);
  unsigned short* catT  = (unsigned short*)alloc((size_t)BATCH * NN * 256 * 2);
  float*          sums  = (float*)alloc(512 * 4);
  if (off > ws_size) return;  // workspace too small: bail deterministically

  // ----- helpers -----
  auto launch_gemm = [&](const unsigned short* A, long long sA, int lda,
                         const unsigned short* Bt, long long sB, int ldb,
                         const float* Zf, const unsigned short* Zh, long long sZ, int ldz,
                         float alpha, float beta, const float* bias,
                         float* D, long long sD, int ldd,
                         unsigned short* Dnat, unsigned short* Dt, long long sDt, int lddt,
                         int M, int Ncols, int Kdim, int nb) {
    // M is always a multiple of 32 here (3072 / 768) -> MT = 2.
    if (Ncols % 64 == 0) {
      dim3 grid(M / 32, Ncols / 64, nb);
      k_gemm<2, 4><<<grid, 32, 0, stream>>>(A, sA, lda, Bt, sB, ldb, Zf, Zh, sZ, ldz,
                                            alpha, beta, bias, D, sD, ldd, Dnat, Dt,
                                            sDt, lddt, Ncols, Kdim);
    } else if (Ncols % 32 == 0) {
      dim3 grid(M / 32, Ncols / 32, nb);
      k_gemm<2, 2><<<grid, 32, 0, stream>>>(A, sA, lda, Bt, sB, ldb, Zf, Zh, sZ, ldz,
                                            alpha, beta, bias, D, sD, ldd, Dnat, Dt,
                                            sDt, lddt, Ncols, Kdim);
    } else {
      dim3 grid(M / 32, (Ncols + 15) / 16, nb);
      k_gemm<2, 1><<<grid, 32, 0, stream>>>(A, sA, lda, Bt, sB, ldb, Zf, Zh, sZ, ldz,
                                            alpha, beta, bias, D, sD, ldd, Dnat, Dt,
                                            sDt, lddt, Ncols, Kdim);
    }
  };

  auto cvt = [&](const float* src, unsigned short* dst, long long n) {
    k_f32_to_bf16<<<(unsigned)((n + 255) / 256), 256, 0, stream>>>(src, dst, n);
  };
  auto cw = [&](const float* W, unsigned short* Wt, int Cin, int Cout, int Pad) {
    long long tot = 3LL * Pad * Cin;
    k_cheb_w<<<(unsigned)((tot + 255) / 256), 256, 0, stream>>>(W, Wt, Cin, Cout, Pad);
  };
  auto bn = [&](const float* p, int nn, int C, const float* gamma, const float* beta_,
                unsigned short* Onat, unsigned short* Ot) {
    k_bn_reduce<<<C, 256, 0, stream>>>(p, BATCH * nn, C, sums);
    long long tot = (long long)BATCH * nn * C;
    k_bn_apply<<<(unsigned)((tot + 255) / 256), 256, 0, stream>>>(
        p, sums, gamma, beta_, BATCH * nn, C, nn, Onat, Ot);
  };
  // cheb_conv: Xnat/XT = input (bf16 both layouts), writes preOut f32 [B][nn][Cout]
  auto cheb = [&](const unsigned short* Xnat, const unsigned short* XT,
                  const unsigned short* Lh, int nn, const unsigned short* Wt,
                  int Cin, int Cout, int Pad, const float* bias, float* preOut) {
    // x1 = L @ x0  (bf16 nat + T outputs)
    launch_gemm(Lh, 0, nn, XT, (long long)Cin * nn, nn,
                nullptr, nullptr, 0, 1, 1.f, 0.f, nullptr,
                nullptr, (long long)nn * Cin, Cin, b1nat, b1T, (long long)Cin * nn, nn,
                nn, Cin, nn, BATCH);
    // x2 = 2*L @ x1 - x0  (Z = x0 from bf16 natural)
    launch_gemm(Lh, 0, nn, b1T, (long long)Cin * nn, nn,
                nullptr, Xnat, (long long)nn * Cin, Cin, 2.f, -1.f, nullptr,
                nullptr, (long long)nn * Cin, Cin, b2nat, nullptr, 0, 1,
                nn, Cin, nn, BATCH);
    // y = x0@W0 + x1@W1 + x2@W2 + bias
    const unsigned short* xs[3] = {Xnat, b1nat, b2nat};
    for (int k = 0; k < 3; ++k) {
      launch_gemm(xs[k], (long long)nn * Cin, Cin,
                  Wt + (size_t)k * Pad * Cin, 0, Cin,
                  k ? preOut : nullptr, nullptr, (long long)nn * Cout, Cout,
                  1.f, k ? 1.f : 0.f, k ? nullptr : bias,
                  preOut, (long long)nn * Cout, Cout, nullptr, nullptr, 0, 1,
                  nn, Cout, Cin, BATCH);
    }
  };

  // ----- precision conversions (L0 bf16 = 18.9MB -> L2-resident) -----
  cvt(L0, L0h, (long long)NN * NN);
  cvt(L1, L1h, (long long)NN2 * NN2);
  cvt(wr1, wr1h, 128 * 32);
  cvt(wr2, wr2h, 128 * 128);
  cvt(uwr1, uwr1h, 64 * 256);
  cw(w11, w11t, 32, 64, 64);     cw(w13, w13t, 64, 128, 128);
  cw(w21, w21t, 128, 192, 192);  cw(w23, w23t, 192, 128, 128);
  cw(uw11, uw11t, 256, 128, 128);cw(uw12, uw12t, 128, 64, 64);
  cw(uw13, uw13t, 64, 8, 16);
  {
    long long tot = (long long)BATCH * NN * 32;
    k_f32_to_natT<<<(unsigned)((tot + 255) / 256), 256, 0, stream>>>(
        x, tot, 32, NN, xnat, xT);
  }

  // ----- encoder level 1 -----
  cheb(xnat, xT, L0h, NN, w11t, 32, 64, 64, b11, pre);
  bn(pre, NN, 64, g11, be11, P0nat, P0T);                       // e1a
  cheb(P0nat, P0T, L0h, NN, w13t, 64, 128, 128, b13, pre);
  bn(pre, NN, 128, g13, be13, P1nat, P1T);                      // e1b
  // e1 = e1b + lin1x1(x, wr1, br1)   (f32 only; consumed by pool + cat)
  launch_gemm(xnat, (long long)NN * 32, 32, wr1h, 0, 32,
              nullptr, P1nat, (long long)NN * 128, 128,
              1.f, 1.f, br1, e1f, (long long)NN * 128, 128,
              nullptr, nullptr, 0, 1, NN, 128, 32, BATCH);
  // pool -> p (bf16 nat+T into P0) + idx
  {
    long long tot = (long long)BATCH * NN2 * 128;
    k_pool<<<(unsigned)((tot + 255) / 256), 256, 0, stream>>>(
        e1f, NN, 128, P0nat, P0T, idx, tot);
  }

  // ----- encoder level 2 (half res) -----
  cheb(P0nat, P0T, L1h, NN2, w21t, 128, 192, 192, b21, pre);
  bn(pre, NN2, 192, g21, be21, P1nat, P1T);                     // e2a
  cheb(P1nat, P1T, L1h, NN2, w23t, 192, 128, 128, b23, pre);
  bn(pre, NN2, 128, g23, be23, P1nat, P1T);                     // e2b (e2a dead)
  // e2 = e2b + lin1x1(p, wr2, br2)
  launch_gemm(P0nat, (long long)NN2 * 128, 128, wr2h, 0, 128,
              nullptr, P1nat, (long long)NN2 * 128, 128,
              1.f, 1.f, br2, e2f, (long long)NN2 * 128, 128,
              nullptr, nullptr, 0, 1, NN2, 128, 128, BATCH);

  // ----- unpool + concat (directly in bf16 nat+T) -----
  {
    long long tot = (long long)BATCH * NN * 256;
    k_cat_init<<<(unsigned)((tot + 255) / 256), 256, 0, stream>>>(
        e1f, NN, catnat, catT, tot);
    long long tot2 = (long long)BATCH * NN2 * 128;
    k_scatter<<<(unsigned)((tot2 + 255) / 256), 256, 0, stream>>>(
        e2f, idx, NN, catnat, catT, tot2);
  }

  // ----- decoder -----
  cheb(catnat, catT, L0h, NN, uw11t, 256, 128, 128, ub11, pre);
  bn(pre, NN, 128, ug11, ube11, P0nat, P0T);                    // d1 (p dead)
  cheb(P0nat, P0T, L0h, NN, uw12t, 128, 64, 64, ub12, pre);
  bn(pre, NN, 64, ug12, ube12, P1nat, P1T);                     // d2 (e2b dead)
  // d = d2 + lin1x1(cat, uwr1, ubr1) -> bf16 nat+T in P0 (d1 dead)
  launch_gemm(catnat, (long long)NN * 256, 256, uwr1h, 0, 256,
              nullptr, P1nat, (long long)NN * 64, 64,
              1.f, 1.f, ubr1, nullptr, (long long)NN * 64, 64,
              P0nat, P0T, (long long)64 * NN, NN, NN, 64, 256, BATCH);
  // final cheb -> d_out f32 [B][N][8] (padded 16-row weights, store guard n<8)
  cheb(P0nat, P0T, L0h, NN, uw13t, 64, 8, 16, ub13, (float*)d_out);
}
